// ComplexAttention_12824772346106
// MI455X (gfx1250) — matrix operand order repaired and checked by reading, not compiled, not run on hardware
//
#include <hip/hip_runtime.h>

// ============================================================================
// ComplexAttention for MI455X (gfx1250, wave32).
// - All matmuls on v_wmma_f32_16x16x32_bf16 (f32 accumulate).
// - All LDS tile staging of bf16 tiles via the Tensor Data Mover
//   (tensor_load_to_lds + s_wait_tensorcnt), with D# LDS padding producing
//   72-element padded rows (bank spread).
// Pipeline:
//   [wtrans] x8                  : W (f32) -> Wt (bf16, transposed [n][k])
//   [gemm64<f32,bf16,rm>] x4     : Qr,Qi,Kr,Ki
//   [gemm64<f32,bf16,cm>] x2     : Vr,Vi written column-major (Vt[d][token])
//   [attn]                       : flash attention + global-token branch
//   [gemm64<bf16,f32,rm>] x2     : out_r @ Wor, out_i @ Woi -> d_out (fp32)
// ============================================================================

typedef __bf16 bf16;
typedef __attribute__((ext_vector_type(16))) __bf16 v16bf;
typedef __attribute__((ext_vector_type(8)))  __bf16 v8bf;
typedef __attribute__((ext_vector_type(8)))  float  v8f;
typedef unsigned int u32x4 __attribute__((ext_vector_type(4)));
typedef int          i32x8 __attribute__((ext_vector_type(8)));
typedef int          i32x4 __attribute__((ext_vector_type(4)));

#define B_   2
#define L_   2048
#define D_   1024
#define H_   16
#define DH_  64
#define G_   64
#define R_   (B_ * L_)          // 4096 rows total
#define SCALE_   0.125f         // DH^-0.5
#define NEG_BIG  (-1.0e30f)
#define LDSPAD   72             // bf16 elements per LDS row (64 + 8 pad)

// A/B fragment: 16 bf16 per lane = 8 VGPRs
union ABFrag {
    v16bf v;
    v8bf  h[2];
};

__device__ __forceinline__ v8f wmma_bf16(const ABFrag& a, const ABFrag& b, v8f c) {
    return __builtin_amdgcn_wmma_f32_16x16x32_bf16(false, a.v, false, b.v,
                                                   (short)0, c, false, false);
}

__device__ __forceinline__ v8f vzero8() {
    v8f z;
#pragma unroll
    for (int e = 0; e < 8; ++e) z[e] = 0.0f;
    return z;
}

// Convert 32 contiguous fp32 -> 32 contiguous bf16 (4x uint4 stores)
__device__ __forceinline__ void cvt_store32(const float* __restrict__ src,
                                            bf16* __restrict__ dst) {
    const float4* s4 = (const float4*)src;
#pragma unroll
    for (int q = 0; q < 4; ++q) {
        float4 a = s4[2 * q], b = s4[2 * q + 1];
        union { bf16 e[8]; uint4 u; } pk;
        pk.e[0] = (bf16)a.x; pk.e[1] = (bf16)a.y;
        pk.e[2] = (bf16)a.z; pk.e[3] = (bf16)a.w;
        pk.e[4] = (bf16)b.x; pk.e[5] = (bf16)b.y;
        pk.e[6] = (bf16)b.z; pk.e[7] = (bf16)b.w;
        *(uint4*)(dst + 8 * q) = pk.u;
    }
}

// ----------------------------------------------------------------------------
// TDM: DMA a 64x64 bf16 tile (row stride = stride_elems) from global memory
// into LDS at byte offset lds_off, padding each 128B row by 16B so the LDS
// image has 72-element (144B) rows.  D# per CDNA5 ISA ch.8; 2D tile so
// descriptor groups 2/3 (and the trailing group) are zero.
// amdgpu-toolchain 6-arg builtin form.
// ----------------------------------------------------------------------------
__device__ __forceinline__ void tdm_load_tile64(const void* gsrc,
                                                unsigned lds_off,
                                                unsigned stride_elems) {
    const unsigned long long ga = (unsigned long long)(size_t)gsrc;
    u32x4 g0;
    g0[0] = 1u;                                        // count=1 (valid, user)
    g0[1] = lds_off;                                   // LDS byte address
    g0[2] = (unsigned)(ga & 0xffffffffu);              // global_addr[31:0]
    g0[3] = (unsigned)((ga >> 32) & 0x01ffffffu)       // global_addr[56:32]
            | (2u << 30);                              // type = 2 ("image")
    i32x8 g1;
    g1[0] = (int)((1u << 16)      // data_size = 1 -> 2 bytes
                | (1u << 20)      // pad_enable
                | (4u << 22)      // pad_interval: 32 DWORDs (128B) per row
                | (3u << 25));    // pad_amount: 4 DWORDs (16B)
    g1[1] = (int)(64u << 16);     // tensor_dim0 = 64  (bits 63:48)
    g1[2] = (int)(64u << 16);     // tensor_dim1 = 64  (bits 95:80)
    g1[3] = (int)(64u << 16);     // tile_dim0   = 64  (bits 127:112)
    g1[4] = 64;                   // tile_dim1   = 64  (bits 143:128)
    g1[5] = (int)stride_elems;    // tensor_dim0_stride[31:0]
    g1[6] = 0;
    g1[7] = 0;
    const i32x4 z4 = {0, 0, 0, 0};
    const i32x8 z8 = {0, 0, 0, 0, 0, 0, 0, 0};
    __builtin_amdgcn_tensor_load_to_lds(g0, g1, z4, z4, z8, 0);
}

__device__ __forceinline__ unsigned lds_offset_of(const void* p) {
    return (unsigned)(size_t)p;   // low 32 bits of generic addr = LDS offset
}

// ============================================================================
// Weight transpose + convert: Wt[n][k] = (bf16)W[k][n]   (one-off, 8 matrices)
// ============================================================================
__global__ __launch_bounds__(128, 1)
void wtrans(const float* __restrict__ W, bf16* __restrict__ Wt) {
    __shared__ alignas(16) bf16 sT[64][LDSPAD];       // [k][n]
    const int tid = threadIdx.x, r = tid >> 1, h2 = tid & 1;
    const int k0 = blockIdx.x * 64, n0 = blockIdx.y * 64;
    cvt_store32(W + (size_t)(k0 + r) * D_ + n0 + h2 * 32, &sT[r][h2 * 32]);
    __syncthreads();
    union { bf16 e[32]; uint4 u[4]; } pk;
#pragma unroll
    for (int j = 0; j < 32; ++j) pk.e[j] = sT[h2 * 32 + j][r];
    uint4* dst = (uint4*)(Wt + (size_t)(n0 + r) * D_ + k0 + h2 * 32);
#pragma unroll
    for (int q = 0; q < 4; ++q) dst[q] = pk.u[q];
}

// ============================================================================
// Tiled GEMM: D[4096x1024] = A[4096x1024] * W[1024x1024]
// Wt is pre-transposed bf16 [n][k] -> staged by TDM.  A staged by TDM when
// already bf16, manual convert when fp32.
// DST_T: write bf16 column-major (Vt[d][token]) for the V projections.
// ============================================================================
template <bool SRC_F32, bool DST_F32, bool DST_T>
__global__ __launch_bounds__(128, 1)
void gemm64(const void* __restrict__ Asrc, const bf16* __restrict__ Wt,
            void* __restrict__ Dst) {
    __shared__ alignas(16) bf16 sA[64][LDSPAD];       // [m][k]
    __shared__ alignas(16) bf16 sW[64][LDSPAD];       // [n][k]

    const int tid = threadIdx.x;
    const int wv = tid >> 5, ln = tid & 31, lh = ln >> 4, l16 = ln & 15;
    const int m0 = blockIdx.x * 64, n0 = blockIdx.y * 64;
    const int srow = tid >> 1, sh = tid & 1;

    const unsigned offA = lds_offset_of(&sA[0][0]);
    const unsigned offW = lds_offset_of(&sW[0][0]);

    v8f acc[4];
#pragma unroll
    for (int s = 0; s < 4; ++s) acc[s] = vzero8();

    for (int k0 = 0; k0 < D_; k0 += 64) {
        if (tid < 32) {                                  // wave 0 drives TDM
            tdm_load_tile64(Wt + (size_t)n0 * D_ + k0, offW, D_);
            if (!SRC_F32)
                tdm_load_tile64((const bf16*)Asrc + (size_t)m0 * D_ + k0,
                                offA, D_);
        }
        if (SRC_F32) {                                   // overlap with DMA
            const float* src = (const float*)Asrc +
                               (size_t)(m0 + srow) * D_ + k0 + sh * 32;
            cvt_store32(src, &sA[srow][sh * 32]);
        }
        if (tid < 32) __builtin_amdgcn_s_wait_tensorcnt(0);
        __syncthreads();
#pragma unroll
        for (int c = 0; c < 2; ++c) {                    // K chunk of 32
            ABFrag a;
            a.h[0] = *(const v8bf*)&sA[wv * 16 + l16][c * 32 + 16 * lh];
            a.h[1] = *(const v8bf*)&sA[wv * 16 + l16][c * 32 + 16 * lh + 8];
#pragma unroll
            for (int s = 0; s < 4; ++s) {                // N sub-tile of 16
                ABFrag b;
                b.h[0] = *(const v8bf*)&sW[s * 16 + l16][c * 32 + 16 * lh];
                b.h[1] = *(const v8bf*)&sW[s * 16 + l16][c * 32 + 16 * lh + 8];
                acc[s] = wmma_bf16(a, b, acc[s]);
            }
        }
        __syncthreads();
    }
#pragma unroll
    for (int s = 0; s < 4; ++s) {
#pragma unroll
        for (int v = 0; v < 8; ++v) {
            const size_t row = (size_t)(m0 + wv * 16 + v + 8 * lh);
            const int col = n0 + s * 16 + l16;
            if (DST_F32)      ((float*)Dst)[row * D_ + col] = acc[s][v];
            else if (DST_T)   ((bf16*)Dst)[(size_t)col * R_ + row] = (bf16)acc[s][v];
            else              ((bf16*)Dst)[row * D_ + col] = (bf16)acc[s][v];
        }
    }
}

// ============================================================================
// Flash attention + global-token branch.
// grid = (L/64, H, B); 128 threads = 4 waves; each wave owns 16 query rows.
// K tiles row-major, V tiles come from column-major Vt -> both TDM-staged.
// ============================================================================
__global__ __launch_bounds__(128, 1)
void attn(const bf16* __restrict__ Qr, const bf16* __restrict__ Qi,
          const bf16* __restrict__ Kr, const bf16* __restrict__ Ki,
          const bf16* __restrict__ Vtr, const bf16* __restrict__ Vti,
          const int* __restrict__ amask,
          const float* __restrict__ gkr, const float* __restrict__ gki,
          const float* __restrict__ gvr, const float* __restrict__ gvi,
          const float* __restrict__ gmix,
          bf16* __restrict__ OutR, bf16* __restrict__ OutI) {
    __shared__ alignas(16) bf16 sKr[64][LDSPAD];   // [key][d]
    __shared__ alignas(16) bf16 sKi[64][LDSPAD];
    __shared__ alignas(16) bf16 sVr[64][LDSPAD];   // [d][key]
    __shared__ alignas(16) bf16 sVi[64][LDSPAD];
    __shared__ alignas(16) bf16 sP[4][16][LDSPAD]; // per-wave P re-layout

    const int tid = threadIdx.x;
    const int wv = tid >> 5, ln = tid & 31, lh = ln >> 4, l16 = ln & 15;
    const int b = blockIdx.z, h = blockIdx.y;
    const int q0 = blockIdx.x * 64 + wv * 16;
    const int srow = tid >> 1, sh = tid & 1;

    const unsigned offKr = lds_offset_of(&sKr[0][0]);
    const unsigned offKi = lds_offset_of(&sKi[0][0]);
    const unsigned offVr = lds_offset_of(&sVr[0][0]);
    const unsigned offVi = lds_offset_of(&sVi[0][0]);

    // ---- preload Q A-fragments ----
    ABFrag aQr[2], aQi[2];
    {
        const size_t base = ((size_t)(b * L_ + q0 + l16)) * D_ + h * DH_;
#pragma unroll
        for (int c = 0; c < 2; ++c) {
            aQr[c].h[0] = *(const v8bf*)(Qr + base + c * 32 + 16 * lh);
            aQr[c].h[1] = *(const v8bf*)(Qr + base + c * 32 + 16 * lh + 8);
            aQi[c].h[0] = *(const v8bf*)(Qi + base + c * 32 + 16 * lh);
            aQi[c].h[1] = *(const v8bf*)(Qi + base + c * 32 + 16 * lh + 8);
        }
    }

    v8f oR[4], oI[4];
#pragma unroll
    for (int s = 0; s < 4; ++s) { oR[s] = vzero8(); oI[s] = vzero8(); }
    float mrun[8], lrun[8];
#pragma unroll
    for (int v = 0; v < 8; ++v) { mrun[v] = NEG_BIG; lrun[v] = 0.0f; }

    // ======================= main key loop (blocks of 64) ==================
    for (int kb = 0; kb < L_ / 64; ++kb) {
        const int key0 = kb * 64;
        if (tid < 32) {                         // wave 0 issues 4 tile DMAs
            const size_t kbase = ((size_t)(b * L_ + key0)) * D_ + h * DH_;
            tdm_load_tile64(Kr + kbase, offKr, D_);
            tdm_load_tile64(Ki + kbase, offKi, D_);
            const size_t vbase = (size_t)(h * DH_) * R_ + b * L_ + key0;
            tdm_load_tile64(Vtr + vbase, offVr, R_);
            tdm_load_tile64(Vti + vbase, offVi, R_);
            __builtin_amdgcn_s_wait_tensorcnt(0);
        }
        __syncthreads();

        // ---- scores: S = Qr*Kr^T + Qi*Ki^T (4 chained WMMAs per sub-tile)
        v8f S[4];
#pragma unroll
        for (int s = 0; s < 4; ++s) {
            v8f a = vzero8();
#pragma unroll
            for (int c = 0; c < 2; ++c) {
                ABFrag bk;
                bk.h[0] = *(const v8bf*)&sKr[s * 16 + l16][c * 32 + 16 * lh];
                bk.h[1] = *(const v8bf*)&sKr[s * 16 + l16][c * 32 + 16 * lh + 8];
                a = wmma_bf16(aQr[c], bk, a);
            }
#pragma unroll
            for (int c = 0; c < 2; ++c) {
                ABFrag bk;
                bk.h[0] = *(const v8bf*)&sKi[s * 16 + l16][c * 32 + 16 * lh];
                bk.h[1] = *(const v8bf*)&sKi[s * 16 + l16][c * 32 + 16 * lh + 8];
                a = wmma_bf16(aQi[c], bk, a);
            }
            S[s] = a;
        }
        // ---- scale + key mask (lane holds one key column) ----
#pragma unroll
        for (int s = 0; s < 4; ++s) {
            const int masked = amask[b * L_ + key0 + s * 16 + l16];
#pragma unroll
            for (int v = 0; v < 8; ++v) {
                const float x = S[s][v] * SCALE_;
                S[s][v] = (masked > 0) ? NEG_BIG : x;
            }
        }
        // ---- online softmax (row reductions over 16-lane groups) ----
        float alpha[8];
#pragma unroll
        for (int v = 0; v < 8; ++v) {
            float mx = fmaxf(fmaxf(S[0][v], S[1][v]), fmaxf(S[2][v], S[3][v]));
#pragma unroll
            for (int off = 1; off < 16; off <<= 1)
                mx = fmaxf(mx, __shfl_xor(mx, off, 32));
            const float mnew = fmaxf(mrun[v], mx);
            alpha[v] = __expf(mrun[v] - mnew);
            mrun[v] = mnew;
            float sum = 0.0f;
#pragma unroll
            for (int s = 0; s < 4; ++s) {
                const float p = __expf(S[s][v] - mnew);
                S[s][v] = p;
                sum += p;
            }
#pragma unroll
            for (int off = 1; off < 16; off <<= 1)
                sum += __shfl_xor(sum, off, 32);
            lrun[v] = lrun[v] * alpha[v] + sum;
        }
        // ---- rescale O, publish P (C layout -> LDS -> A layout) ----
#pragma unroll
        for (int s = 0; s < 4; ++s)
#pragma unroll
            for (int v = 0; v < 8; ++v) {
                oR[s][v] *= alpha[v];
                oI[s][v] *= alpha[v];
                sP[wv][v + 8 * lh][s * 16 + l16] = (bf16)S[s][v];
            }
        __syncthreads();
        // ---- O += P @ V ----
#pragma unroll
        for (int c = 0; c < 2; ++c) {
            ABFrag ap;
            ap.h[0] = *(const v8bf*)&sP[wv][l16][c * 32 + 16 * lh];
            ap.h[1] = *(const v8bf*)&sP[wv][l16][c * 32 + 16 * lh + 8];
#pragma unroll
            for (int s = 0; s < 4; ++s) {
                ABFrag bv;
                bv.h[0] = *(const v8bf*)&sVr[s * 16 + l16][c * 32 + 16 * lh];
                bv.h[1] = *(const v8bf*)&sVr[s * 16 + l16][c * 32 + 16 * lh + 8];
                oR[s] = wmma_bf16(ap, bv, oR[s]);
                bv.h[0] = *(const v8bf*)&sVi[s * 16 + l16][c * 32 + 16 * lh];
                bv.h[1] = *(const v8bf*)&sVi[s * 16 + l16][c * 32 + 16 * lh + 8];
                oI[s] = wmma_bf16(ap, bv, oI[s]);
            }
        }
        __syncthreads();
    }
    // ---- normalize main attention output ----
#pragma unroll
    for (int v = 0; v < 8; ++v) {
        const float inv = 1.0f / fmaxf(lrun[v], 1e-20f);
#pragma unroll
        for (int s = 0; s < 4; ++s) { oR[s][v] *= inv; oI[s][v] *= inv; }
    }

    // ================= global learned-token branch (G = 64) ================
    {
        cvt_store32(gkr + ((size_t)h * G_ + srow) * DH_ + sh * 32,
                    &sKr[srow][sh * 32]);
        cvt_store32(gki + ((size_t)h * G_ + srow) * DH_ + sh * 32,
                    &sKi[srow][sh * 32]);
        {
            const float4* gr4 = (const float4*)(gvr + ((size_t)h * G_ + srow) * DH_ + sh * 32);
            const float4* gi4 = (const float4*)(gvi + ((size_t)h * G_ + srow) * DH_ + sh * 32);
#pragma unroll
            for (int q = 0; q < 8; ++q) {
                float4 fr = gr4[q], fi = gi4[q];
                const int d0 = sh * 32 + q * 4;
                sVr[d0 + 0][srow] = (bf16)fr.x; sVi[d0 + 0][srow] = (bf16)fi.x;
                sVr[d0 + 1][srow] = (bf16)fr.y; sVi[d0 + 1][srow] = (bf16)fi.y;
                sVr[d0 + 2][srow] = (bf16)fr.z; sVi[d0 + 2][srow] = (bf16)fi.z;
                sVr[d0 + 3][srow] = (bf16)fr.w; sVi[d0 + 3][srow] = (bf16)fi.w;
            }
        }
        __syncthreads();

        v8f Sg[4];
#pragma unroll
        for (int s = 0; s < 4; ++s) {
            v8f a = vzero8();
#pragma unroll
            for (int c = 0; c < 2; ++c) {
                ABFrag bk;
                bk.h[0] = *(const v8bf*)&sKr[s * 16 + l16][c * 32 + 16 * lh];
                bk.h[1] = *(const v8bf*)&sKr[s * 16 + l16][c * 32 + 16 * lh + 8];
                a = wmma_bf16(aQr[c], bk, a);
            }
#pragma unroll
            for (int c = 0; c < 2; ++c) {
                ABFrag bk;
                bk.h[0] = *(const v8bf*)&sKi[s * 16 + l16][c * 32 + 16 * lh];
                bk.h[1] = *(const v8bf*)&sKi[s * 16 + l16][c * 32 + 16 * lh + 8];
                a = wmma_bf16(aQi[c], bk, a);
            }
            Sg[s] = a;
        }
        const float mix = 1.0f / (1.0f + __expf(-gmix[0]));
#pragma unroll
        for (int v = 0; v < 8; ++v) {
            float mx = fmaxf(fmaxf(Sg[0][v] * SCALE_, Sg[1][v] * SCALE_),
                             fmaxf(Sg[2][v] * SCALE_, Sg[3][v] * SCALE_));
#pragma unroll
            for (int off = 1; off < 16; off <<= 1)
                mx = fmaxf(mx, __shfl_xor(mx, off, 32));
            float sum = 0.0f;
#pragma unroll
            for (int s = 0; s < 4; ++s) {
                const float p = __expf(Sg[s][v] * SCALE_ - mx);
                Sg[s][v] = p;
                sum += p;
            }
#pragma unroll
            for (int off = 1; off < 16; off <<= 1)
                sum += __shfl_xor(sum, off, 32);
            const float coef = mix / sum;
#pragma unroll
            for (int s = 0; s < 4; ++s)
                sP[wv][v + 8 * lh][s * 16 + l16] = (bf16)(Sg[s][v] * coef);
        }
        __syncthreads();
#pragma unroll
        for (int c = 0; c < 2; ++c) {
            ABFrag ap;
            ap.h[0] = *(const v8bf*)&sP[wv][l16][c * 32 + 16 * lh];
            ap.h[1] = *(const v8bf*)&sP[wv][l16][c * 32 + 16 * lh + 8];
#pragma unroll
            for (int s = 0; s < 4; ++s) {
                ABFrag bv;
                bv.h[0] = *(const v8bf*)&sVr[s * 16 + l16][c * 32 + 16 * lh];
                bv.h[1] = *(const v8bf*)&sVr[s * 16 + l16][c * 32 + 16 * lh + 8];
                oR[s] = wmma_bf16(ap, bv, oR[s]);
                bv.h[0] = *(const v8bf*)&sVi[s * 16 + l16][c * 32 + 16 * lh];
                bv.h[1] = *(const v8bf*)&sVi[s * 16 + l16][c * 32 + 16 * lh + 8];
                oI[s] = wmma_bf16(ap, bv, oI[s]);
            }
        }
    }

    // ---- write merged outputs (bf16) for the final projections ----
#pragma unroll
    for (int s = 0; s < 4; ++s)
#pragma unroll
        for (int v = 0; v < 8; ++v) {
            const size_t row = (size_t)(b * L_ + q0 + v + 8 * lh);
            const int col = h * DH_ + s * 16 + l16;
            OutR[row * D_ + col] = (bf16)oR[s][v];
            OutI[row * D_ + col] = (bf16)oI[s][v];
        }
}

// ============================================================================
extern "C" void kernel_launch(void* const* d_in, const int* in_sizes, int n_in,
                              void* d_out, int out_size, void* d_ws, size_t ws_size,
                              hipStream_t stream) {
    (void)in_sizes; (void)n_in; (void)out_size; (void)ws_size;

    const float* r    = (const float*)d_in[0];
    const float* im   = (const float*)d_in[1];
    const int*   msk  = (const int*)d_in[2];
    const float* Wsrc[8] = {
        (const float*)d_in[3], (const float*)d_in[4],   // Wqr, Wqi
        (const float*)d_in[5], (const float*)d_in[6],   // Wkr, Wki
        (const float*)d_in[7], (const float*)d_in[8],   // Wvr, Wvi
        (const float*)d_in[9], (const float*)d_in[10],  // Wor, Woi
    };
    const float* gkr  = (const float*)d_in[11];
    const float* gki  = (const float*)d_in[12];
    const float* gvr  = (const float*)d_in[13];
    const float* gvi  = (const float*)d_in[14];
    const float* gmix = (const float*)d_in[15];

    const size_t RD = (size_t)R_ * D_;
    const size_t DD = (size_t)D_ * D_;
    bf16* ws = (bf16*)d_ws;                 // 8*RD + 8*DD bf16 = 80 MB
    bf16 *Qr  = ws + 0 * RD, *Qi  = ws + 1 * RD;
    bf16 *Kr  = ws + 2 * RD, *Ki  = ws + 3 * RD;
    bf16 *Vtr = ws + 4 * RD, *Vti = ws + 5 * RD;     // column-major [D][R]
    bf16 *Omr = ws + 6 * RD, *Omi = ws + 7 * RD;
    bf16 *Wt  = ws + 8 * RD;                          // 8 transposed weights

    const dim3 gblk(128);

    // one-off weight transpose + bf16 convert
    for (int w = 0; w < 8; ++w)
        wtrans<<<dim3(D_ / 64, D_ / 64), gblk, 0, stream>>>(Wsrc[w], Wt + w * DD);

    const dim3 ggrid(R_ / 64, D_ / 64);     // 64 x 16
    // QKV projections
    gemm64<true, false, false><<<ggrid, gblk, 0, stream>>>(r,  Wt + 0 * DD, Qr);
    gemm64<true, false, false><<<ggrid, gblk, 0, stream>>>(im, Wt + 1 * DD, Qi);
    gemm64<true, false, false><<<ggrid, gblk, 0, stream>>>(r,  Wt + 2 * DD, Kr);
    gemm64<true, false, false><<<ggrid, gblk, 0, stream>>>(im, Wt + 3 * DD, Ki);
    gemm64<true, false, true ><<<ggrid, gblk, 0, stream>>>(r,  Wt + 4 * DD, Vtr);
    gemm64<true, false, true ><<<ggrid, gblk, 0, stream>>>(im, Wt + 5 * DD, Vti);

    // flash attention + global branch
    attn<<<dim3(L_ / 64, H_, B_), gblk, 0, stream>>>(
        Qr, Qi, Kr, Ki, Vtr, Vti, msk, gkr, gki, gvr, gvi, gmix, Omr, Omi);

    // output projections -> fp32 d_out (real then imag)
    float* outR = (float*)d_out;
    float* outI = outR + RD;
    gemm64<false, true, false><<<ggrid, gblk, 0, stream>>>(Omr, Wt + 6 * DD, outR);
    gemm64<false, true, false><<<ggrid, gblk, 0, stream>>>(Omi, Wt + 7 * DD, outI);
}
